// MultiHeadAttention_52527450030314
// MI455X (gfx1250) — compile-verified, matrix operand order checked by reading, not run on hardware
//
#include <hip/hip_runtime.h>
#include <hip/hip_bf16.h>
#include <math.h>

typedef __attribute__((ext_vector_type(16))) _Float16 v16h;
typedef __attribute__((ext_vector_type(8)))  _Float16 v8h;
typedef __attribute__((ext_vector_type(4)))  _Float16 v4h;
typedef __attribute__((ext_vector_type(8)))  float    v8f;

#define D_MODEL   512
#define NUM_HEADS 6
#define MIN_DIM   8
#define B_SZ      4
#define T_SZ      2048
#define NROW      (B_SZ * T_SZ)      // 8192
#define QKV_N     (3 * D_MODEL)      // 1536
#define PADW      704                // >= 512 + 6*31 rounded to 32; fixed stride

// meta layout (ints): [0..5] dim, [6..12] start (start[6]=512), [13..19] pstart
#define META_START  6
#define META_PSTART 13

__device__ __forceinline__ v8f wmma_f16(v16h a, v16h b, v8f c) {
  return __builtin_amdgcn_wmma_f32_16x16x32_f16(false, a, false, b,
                                                (short)0, c, false, false);
}

// ---------------------------------------------------------------------------
// Unified fragment loader from row-major f16, NO guards, two b128 loads.
// A-frag:  P = A (MxK row-major),  row0 = tile row,  lane row = row0+(lane&15)
// B-frag:  P = B^T (NxK row-major), row0 = tile col (B[k][n] = P[n][k])
// Element 2v+j of the fragment = P[row][k0 + (v>>2)*16 + half*8 + (v&3)*2 + j].
// Works for LDS pointers too (addrspace inference -> ds_load_b128).
// ---------------------------------------------------------------------------
__device__ __forceinline__ v16h load_frag(const _Float16* __restrict__ P, int ld,
                                          int row0, int k0) {
  const int lane = threadIdx.x & 31;
  const int half = lane >> 4;
  const _Float16* base = P + (size_t)(row0 + (lane & 15)) * ld + k0 + half * 8;
  v8h lo = *(const v8h*)(base);        // grp 0: K = k0 + half*8 + 0..7
  v8h hi = *(const v8h*)(base + 16);   // grp 1: K = k0 + 16 + half*8 + 0..7
  v16h out;
#pragma unroll
  for (int i = 0; i < 8; ++i) { out[i] = lo[i]; out[8 + i] = hi[i]; }
  return out;
}

// ---------------------------------------------------------------------------
// Async DMA of one 32x32 f16 tile (rows [s0,s0+32) x cols [kg,kg+32)) from a
// PADW-stride f16 matrix into LDS (row-major, ld=32). 4 issues x 32 lanes x
// 16B = 2KB, tracked by ASYNCcnt (overlaps WMMA instead of blocking LOADcnt).
// ---------------------------------------------------------------------------
__device__ __forceinline__ void async_stage_tile(const _Float16* __restrict__ G,
                                                 int s0, int kg, unsigned lds_base) {
  const int lane = threadIdx.x & 31;
  const int row  = lane >> 2;          // 0..7 (+8 per issue)
  const int colb = (lane & 3) * 16;    // byte offset within 64B row
#pragma unroll
  for (int j = 0; j < 4; ++j) {
    const void* g = (const char*)(G + (size_t)(s0 + row + j * 8) * PADW + kg) + colb;
    unsigned    l = lds_base + (unsigned)((row + j * 8) * 64 + colb);
    asm volatile("global_load_async_to_lds_b128 %0, %1, off"
                 :: "v"(l), "v"(g) : "memory");
  }
}

// ---------------------------------------------------------------------------
// K0: head-dim rounding + loss + meta tables (single thread; trivial).
// ---------------------------------------------------------------------------
__global__ void dims_loss_kernel(const float* __restrict__ logits,
                                 int* __restrict__ meta,
                                 float* __restrict__ loss_out) {
  if (threadIdx.x != 0 || blockIdx.x != 0) return;
  float p[NUM_HEADS], hdf[NUM_HEADS], res[NUM_HEADS];
  float mx = logits[0];
  for (int i = 1; i < NUM_HEADS; ++i) mx = fmaxf(mx, logits[i]);
  float se = 0.0f;
  for (int i = 0; i < NUM_HEADS; ++i) { p[i] = __expf(logits[i] - mx); se += p[i]; }
  long hdi[NUM_HEADS]; long isum = 0; float fsum = 0.0f;
  for (int i = 0; i < NUM_HEADS; ++i) {
    p[i] /= se;
    hdf[i] = (float)MIN_DIM + p[i] * (float)(D_MODEL - MIN_DIM * NUM_HEADS);
    fsum += hdf[i];
    hdi[i] = (long)floorf(hdf[i]);
    res[i] = hdf[i] - (float)hdi[i];
    isum += hdi[i];
  }
  int diff = (int)((long)D_MODEL - isum);
  bool used[NUM_HEADS] = {false, false, false, false, false, false};
  if (diff > 0) {
    for (int d = 0; d < diff; ++d) {
      int best = -1; float bv = -1.0f;
      for (int i = 0; i < NUM_HEADS; ++i)
        if (!used[i] && res[i] > bv) { bv = res[i]; best = i; }
      used[best] = true; hdi[best] += 1;
    }
  } else if (diff < 0) {
    for (int d = 0; d < -diff; ++d) {
      int best = -1; float bv = 2.0f;
      for (int i = 0; i < NUM_HEADS; ++i)
        if (!used[i] && res[i] < bv) { bv = res[i]; best = i; }
      used[best] = true; hdi[best] -= 1;
    }
  }
  float fl = (fsum - (float)D_MODEL) * (fsum - (float)D_MODEL);
  float relu_sum = 0.0f, int_sum = 0.0f;
  int s = 0, ps = 0;
  for (int i = 0; i < NUM_HEADS; ++i) {
    relu_sum += fmaxf((float)MIN_DIM - hdf[i], 0.0f);
    float d = hdf[i] - (float)hdi[i];
    int_sum += d * d;
    meta[i] = (int)hdi[i];
    meta[META_START + i]  = s;
    meta[META_PSTART + i] = ps;
    s  += (int)hdi[i];
    ps += ((int)hdi[i] + 31) & ~31;
  }
  meta[META_START + NUM_HEADS]  = s;    // == 512
  meta[META_PSTART + NUM_HEADS] = ps;   // padded total (<= PADW)
  *loss_out = fl + relu_sum / (float)NUM_HEADS + 0.5f * (int_sum / (float)NUM_HEADS);
}

// ---------------------------------------------------------------------------
// Elementwise helpers (bandwidth-bound pre-pass, ~2us at 23.3 TB/s).
// ---------------------------------------------------------------------------
__global__ void zero_f16x8(_Float16* __restrict__ p, int n8) {
  int i = blockIdx.x * blockDim.x + threadIdx.x;
  if (i < n8) { v8h z = {}; ((v8h*)p)[i] = z; }
}

__global__ void cvt_f16x4(const float* __restrict__ in, _Float16* __restrict__ out, int n4) {
  int i = blockIdx.x * blockDim.x + threadIdx.x;
  if (i < n4) {
    float4 v = ((const float4*)in)[i];
    v4h o = { (_Float16)v.x, (_Float16)v.y, (_Float16)v.z, (_Float16)v.w };
    ((v4h*)out)[i] = o;
  }
}

// out[n][k] = (f16) in[k][n]   (in: KxN f32 row-major)
__global__ void transpose_cvt(const float* __restrict__ in, _Float16* __restrict__ out,
                              int K, int N) {
  int n = blockIdx.x * 16 + threadIdx.x;
  int k = blockIdx.y * 16 + threadIdx.y;
  if (n < N && k < K) out[(size_t)n * K + k] = (_Float16)in[(size_t)k * N + n];
}

// ---------------------------------------------------------------------------
// K1: QKV GEMM. C = qh(8192x512) @ wqT^T + b_qkv, epilogue scatters f16 into
// padded per-head Qp/Kp (row-major, stride PADW) and Vt (transposed, stride T).
// ---------------------------------------------------------------------------
__global__ void __launch_bounds__(32)
gemm_qkv(const _Float16* __restrict__ A, const _Float16* __restrict__ Bt,
         const float* __restrict__ bias,
         _Float16* __restrict__ Qp, _Float16* __restrict__ Kp,
         _Float16* __restrict__ Vt, const int* __restrict__ meta) {
  const int n0   = blockIdx.x * 16;
  const int m0   = blockIdx.y * 16;
  const int lane = threadIdx.x & 31;
  const int half = lane >> 4;
  v8f acc = {};
#pragma unroll 4
  for (int k0 = 0; k0 < D_MODEL; k0 += 32) {
    v16h a = load_frag(A,  D_MODEL, m0, k0);
    v16h b = load_frag(Bt, D_MODEL, n0, k0);
    acc = wmma_f16(a, b, acc);
  }
  const int c   = n0 + (lane & 15);        // 0..1535
  const int seg = c >> 9;                  // 0=Q 1=K 2=V (uniform per block)
  const int d   = c & (D_MODEL - 1);
  const int* start  = meta + META_START;
  const int* pstart = meta + META_PSTART;
  int h = 0;
#pragma unroll
  for (int i = 0; i < NUM_HEADS - 1; ++i) h += (d >= start[i + 1]) ? 1 : 0;
  const int  pd = pstart[h] + (d - start[h]);
  const float bv = bias[c];
#pragma unroll
  for (int r = 0; r < 8; ++r) {
    const int row = m0 + r + 8 * half;     // flat b*T + t
    _Float16 val = (_Float16)(acc[r] + bv);
    if (seg == 0)      Qp[(size_t)row * PADW + pd] = val;
    else if (seg == 1) Kp[(size_t)row * PADW + pd] = val;
    else {
      const int b = row >> 11, t = row & (T_SZ - 1);
      Vt[((size_t)b * PADW + pd) * T_SZ + t] = val;
    }
  }
}

// ---------------------------------------------------------------------------
// K2: flash attention. One wave = (batch, head, 16 query rows, 64 padded dim
// cols). Key tiles of 32, DMA'd into double-buffered LDS with
// GLOBAL_LOAD_ASYNC_TO_LDS_B128 so the copy of tile n+1 overlaps the WMMA +
// softmax of tile n (ASYNCcnt, in-order completion -> wait <=4 drains the
// current tile). Zero padding keeps every fragment load unconditional.
// ---------------------------------------------------------------------------
__global__ void __launch_bounds__(32)
flash_attn_kernel(const _Float16* __restrict__ Qp, const _Float16* __restrict__ Kp,
                  const _Float16* __restrict__ Vt, _Float16* __restrict__ attn,
                  const int* __restrict__ meta) {
  __shared__ __align__(16) _Float16 ldsP[16 * 32];
  __shared__ __align__(16) _Float16 ldsK[2][32 * 32];

  const int t0  = blockIdx.x * 16;
  const int bh  = blockIdx.y;
  const int b   = bh / NUM_HEADS;
  const int h   = bh % NUM_HEADS;
  const int dim    = meta[h];
  const int start  = meta[META_START + h];
  const int pstart = meta[META_PSTART + h];
  const int pdim   = (dim + 31) & ~31;
  const int cbase  = blockIdx.z * 64;
  if (cbase >= pdim) return;                      // wave-uniform

  const int lane = threadIdx.x & 31;
  const int half = lane >> 4;

  const _Float16* Qb = Qp + (size_t)b * T_SZ * PADW;
  const _Float16* Kb = Kp + (size_t)b * T_SZ * PADW;
  const _Float16* Vb = Vt + (size_t)b * PADW * T_SZ;

  const float scale = rsqrtf((float)dim);

  float m[8], l[8];
  v8f o[4] = {};
#pragma unroll
  for (int r = 0; r < 8; ++r) { m[r] = -1e30f; l[r] = 0.0f; }

  const int nk    = pdim >> 5;                    // k-chunks per S tile
  const int total = (T_SZ / 32) * nk;             // flattened (s0, kc) count
  const unsigned kbuf0 = (unsigned)(size_t)&ldsK[0][0];
  const unsigned kbuf1 = (unsigned)(size_t)&ldsK[1][0];

  async_stage_tile(Kb, 0, pstart, kbuf0);         // prime the pipeline
  int buf = 0, idx = 0;

  for (int s0 = 0; s0 < T_SZ; s0 += 32) {
    if (s0 + 32 < T_SZ)                           // global_prefetch_b8 (V path)
      __builtin_prefetch(Vb + ((size_t)pstart + cbase) * T_SZ + s0 + 32, 0, 1);

    // S (16x32) = Q @ K^T, K tiles streamed through LDS.
    v8f sA = {}, sB = {};
    for (int kc = 0; kc < nk; ++kc, ++idx) {
      const int nidx = idx + 1;
      if (nidx < total) {                         // uniform
        // our ds_load reads of the other buffer (2 tiles ago) must be done
        asm volatile("s_wait_dscnt 0x0" ::: "memory");
        const int ns0 = (nidx / nk) * 32;
        const int nkg = (nidx % nk) * 32;
        async_stage_tile(Kb, ns0, pstart + nkg, buf ? kbuf0 : kbuf1);
        asm volatile("s_wait_asynccnt 0x4" ::: "memory"); // current tile landed
      } else {
        asm volatile("s_wait_asynccnt 0x0" ::: "memory");
      }
      v16h qf  = load_frag(Qb, PADW, t0, pstart + kc * 32);
      v16h kf0 = load_frag(&ldsK[buf][0], 32, 0,  0);  // keys s0..s0+15
      v16h kf1 = load_frag(&ldsK[buf][0], 32, 16, 0);  // keys s0+16..s0+31
      sA = wmma_f16(qf, kf0, sA);
      sB = wmma_f16(qf, kf1, sB);
      buf ^= 1;
    }

    // Online softmax; each row lives across one 16-lane half.
#pragma unroll
    for (int r = 0; r < 8; ++r) {
      float sa = sA[r] * scale;
      float sb = sB[r] * scale;
      float mxv = fmaxf(sa, sb);
#pragma unroll
      for (int off = 8; off >= 1; off >>= 1)
        mxv = fmaxf(mxv, __shfl_xor(mxv, off, 32));
      float mn = fmaxf(m[r], mxv);
      float pa = __expf(sa - mn);
      float pb = __expf(sb - mn);
      float cf = __expf(m[r] - mn);
      float rs = pa + pb;
#pragma unroll
      for (int off = 8; off >= 1; off >>= 1)
        rs += __shfl_xor(rs, off, 32);
      l[r] = l[r] * cf + rs;
      m[r] = mn;
#pragma unroll
      for (int nc = 0; nc < 4; ++nc) o[nc][r] *= cf;
      const int prow = r + 8 * half;              // C/D-layout row
      ldsP[prow * 32 + (lane & 15)]      = (_Float16)pa;
      ldsP[prow * 32 + 16 + (lane & 15)] = (_Float16)pb;
    }
    asm volatile("s_wait_dscnt 0x0" ::: "memory");

    // P as A-fragment: two ds_load_b128 per lane.
    v16h pf;
    {
      const _Float16* base = ldsP + (lane & 15) * 32 + half * 8;
      v8h lo = *(const v8h*)(base);
      v8h hi = *(const v8h*)(base + 16);
#pragma unroll
      for (int i = 0; i < 8; ++i) { pf[i] = lo[i]; pf[8 + i] = hi[i]; }
    }

    // O += P @ V (V pre-transposed: B-frag == row-major frag of Vt).
#pragma unroll
    for (int nc = 0; nc < 4; ++nc) {
      if (cbase + nc * 16 < pdim) {               // uniform guard
        v16h vf = load_frag(Vb, T_SZ, pstart + cbase + nc * 16, s0);
        o[nc] = wmma_f16(pf, vf, o[nc]);
      }
    }
  }

  // Normalize; write f16 context at the unpadded concat position.
#pragma unroll
  for (int nc = 0; nc < 4; ++nc) {
    const int col = cbase + nc * 16 + (lane & 15);
    if (col < dim) {
#pragma unroll
      for (int r = 0; r < 8; ++r) {
        const int row = t0 + r + 8 * half;
        attn[((size_t)b * T_SZ + row) * D_MODEL + start + col] =
            (_Float16)(o[nc][r] / l[r]);
      }
    }
  }
}

// ---------------------------------------------------------------------------
// K3: output projection. C(8192x512 f32) = attn(f16) @ woT^T + b_out.
// ---------------------------------------------------------------------------
__global__ void __launch_bounds__(32)
gemm_out(const _Float16* __restrict__ A, const _Float16* __restrict__ Bt,
         const float* __restrict__ bias, float* __restrict__ C) {
  const int n0   = blockIdx.x * 16;
  const int m0   = blockIdx.y * 16;
  const int lane = threadIdx.x & 31;
  const int half = lane >> 4;
  v8f acc = {};
#pragma unroll 4
  for (int k0 = 0; k0 < D_MODEL; k0 += 32) {
    v16h a = load_frag(A,  D_MODEL, m0, k0);
    v16h b = load_frag(Bt, D_MODEL, n0, k0);
    acc = wmma_f16(a, b, acc);
  }
  const int col = n0 + (lane & 15);
  const float bv = bias[col];
#pragma unroll
  for (int r = 0; r < 8; ++r) {
    const int row = m0 + r + 8 * half;
    C[(size_t)row * D_MODEL + col] = acc[r] + bv;
  }
}

// ---------------------------------------------------------------------------
extern "C" void kernel_launch(void* const* d_in, const int* in_sizes, int n_in,
                              void* d_out, int out_size, void* d_ws, size_t ws_size,
                              hipStream_t stream) {
  const float* query  = (const float*)d_in[0];
  const float* logits = (const float*)d_in[1];
  const float* w_qkv  = (const float*)d_in[2];
  const float* b_qkv  = (const float*)d_in[3];
  const float* w_out  = (const float*)d_in[4];
  const float* b_out  = (const float*)d_in[5];
  float* out = (float*)d_out;

  // Workspace (f16 unless noted):
  _Float16* qh   = (_Float16*)d_ws;                       // 8192 x 512
  _Float16* wqT  = qh  + (size_t)NROW * D_MODEL;          // 1536 x 512 (transposed)
  _Float16* woT  = wqT + (size_t)QKV_N * D_MODEL;         //  512 x 512 (transposed)
  _Float16* Qp   = woT + (size_t)D_MODEL * D_MODEL;       // 8192 x PADW
  _Float16* Kp   = Qp  + (size_t)NROW * PADW;             // 8192 x PADW
  _Float16* Vt   = Kp  + (size_t)NROW * PADW;             // B * PADW x T
  _Float16* attn = Vt  + (size_t)B_SZ * PADW * T_SZ;      // 8192 x 512
  int*      meta = (int*)(attn + (size_t)NROW * D_MODEL); // 20 ints

  dims_loss_kernel<<<1, 32, 0, stream>>>(logits, meta, out + (out_size - 1));

  // Zero the padded Q/K/V region (contiguous: 3 * 8192 * PADW f16).
  {
    int n8 = (3 * NROW * PADW) / 8;
    zero_f16x8<<<(n8 + 255) / 256, 256, 0, stream>>>(Qp, n8);
  }
  // f32 -> f16 conversions / weight transposes.
  {
    int n4 = (NROW * D_MODEL) / 4;
    cvt_f16x4<<<(n4 + 255) / 256, 256, 0, stream>>>(query, qh, n4);
  }
  transpose_cvt<<<dim3(QKV_N / 16, D_MODEL / 16), dim3(16, 16), 0, stream>>>(
      w_qkv, wqT, D_MODEL, QKV_N);
  transpose_cvt<<<dim3(D_MODEL / 16, D_MODEL / 16), dim3(16, 16), 0, stream>>>(
      w_out, woT, D_MODEL, D_MODEL);

  gemm_qkv<<<dim3(QKV_N / 16, NROW / 16), 32, 0, stream>>>(
      qh, wqT, b_qkv, Qp, Kp, Vt, meta);

  flash_attn_kernel<<<dim3(T_SZ / 16, B_SZ * NUM_HEADS, 8), 32, 0, stream>>>(
      Qp, Kp, Vt, attn, meta);

  gemm_out<<<dim3(D_MODEL / 16, NROW / 16), 32, 0, stream>>>(
      attn, woT, b_out, out);
}
